// Colorizer_30064771072907
// MI455X (gfx1250) — matrix operand order repaired and checked by reading, not compiled, not run on hardware
//
#include <hip/hip_runtime.h>
#include <hip/hip_bf16.h>
#include <math.h>

typedef __attribute__((ext_vector_type(16))) __bf16 v16bf;
typedef __attribute__((ext_vector_type(8)))  float  v8f;

static constexpr int kH   = 48;
static constexpr int kW   = 48;
static constexpr int kC   = 64;     // feature channels
static constexpr int kQC  = 32;     // quantized channels
static constexpr int kR   = 12;
static constexpr int kP   = 25;     // 2R+1
static constexpr int kN   = 625;    // P*P
static constexpr int kK   = 20;     // top-k
static constexpr int kNP  = kH * kW;  // 2304
static constexpr int kNK  = 60;     // 3 refs * 20
#define NEG_INF (-3.402823466e38f)

__device__ __forceinline__ unsigned short f2bf(float f) {
  union { float f; unsigned u; } x; x.f = f;
  unsigned r = x.u + 0x7FFFu + ((x.u >> 16) & 1u);   // round-to-nearest-even
  return (unsigned short)(r >> 16);
}

union FragBF { unsigned u[8]; v16bf v; };

// ---------------------------------------------------------------------------
// Correlation via WMMA. One workgroup = one row y, 16 x-positions.
// For each dy (25 iters): corr(x, dx) = [A(16x64) x B(64xNCOL)](x, n=x+DIL*dx),
// A = ft tile (bf16), B = fr row slice (bf16, zero-padded, stored transposed
// [n][k] so fragment VGPRs load as ds_load_b128).  D elements scatter through
// an address-select (invalid -> LDS trash word) so stores stay unpredicated.
// mode 0: per-position top-20 -> (vals, idx).  mode 1: soft-argmax -> (offy, offx).
// ---------------------------------------------------------------------------
template <int DIL>
__global__ __launch_bounds__(256)
void corr_topk_kernel(const float* __restrict__ ft, const float* __restrict__ fr,
                      int mode,
                      float* __restrict__ out_vals, int* __restrict__ out_idx,
                      float* __restrict__ out_offy, float* __restrict__ out_offx) {
  constexpr int ntiles = (16 + 24 * DIL + 15) >> 4;   // DIL=1 -> 3, DIL=2 -> 4
  constexpr int ncol   = ntiles * 16;

  __shared__ unsigned short sA [16][kC];     // [m][k]  2 KB
  __shared__ unsigned short sBt[ncol][kC];   // [n][k]  6/8 KB (transposed!)
  __shared__ float scorr[16][kN];            // 40 KB
  __shared__ float strash;                   // dump slot for invalid D elements

  const int tid  = threadIdx.x;
  const int lane = tid & 31;                 // wave32
  const int wave = tid >> 5;                 // 8 waves
  const int x0   = blockIdx.x * 16;
  const int y    = blockIdx.y;
  const int hi16 = (lane >= 16);

  // load A tile: A[m][k] = ft[k, y, x0+m]
  for (int i = tid; i < 16 * kC; i += 256) {
    int m = i >> 6, k = i & 63;
    sA[m][k] = f2bf(ft[k * kNP + y * kW + x0 + m]);
  }
  __syncthreads();

  // build A fragments (16-bit A 16x32 layout, ISA 7.12.2) as dword LDS loads
  FragBF a0, a1;
  {
    const unsigned short* rowA = sA[lane & 15];
    #pragma unroll
    for (int vv = 0; vv < 8; ++vv) {
      int kl = ((vv < 4) ? (2 * vv) : (16 + 2 * (vv - 4))) + (hi16 ? 8 : 0);
      a0.u[vv] = *(const unsigned*)(rowA + kl);
      a1.u[vv] = *(const unsigned*)(rowA + 32 + kl);
    }
  }

  for (int dyi = 0; dyi < kP; ++dyi) {
    int yrow = y + (dyi - kR) * DIL;
    // load B transposed: sBt[n][k], column xcol = x0 - R*DIL + n, zero-padded
    for (int i = tid; i < kC * ncol; i += 256) {
      int cch = i / ncol, n = i - cch * ncol;       // consecutive tid -> consecutive xcol
      int xcol = x0 - kR * DIL + n;
      float v = 0.f;
      if (yrow >= 0 && yrow < kH && xcol >= 0 && xcol < kW)
        v = fr[cch * kNP + yrow * kW + xcol];
      sBt[n][cch] = f2bf(v);
    }
    // prefetch next dy row while this tile is consumed (global_prefetch_b8)
    if (dyi + 1 < kP) {
      int yn = y + (dyi + 1 - kR) * DIL;
      if (yn >= 0 && yn < kH) {
        int xc = x0 - kR * DIL; if (xc < 0) xc = 0;
        __builtin_prefetch(fr + (tid & 63) * kNP + yn * kW + xc, 0, 1);
      }
    }
    __syncthreads();

    if (wave < ntiles) {   // wave-uniform branch: EXEC stays all ones for WMMA
      const int n = (wave << 4) + (lane & 15);
      FragBF b0, b1;
      {
        const unsigned short* rowB = sBt[n];
        #pragma unroll
        for (int vv = 0; vv < 8; ++vv) {
          int kl = ((vv < 4) ? (2 * vv) : (16 + 2 * (vv - 4))) + (hi16 ? 8 : 0);
          b0.u[vv] = *(const unsigned*)(rowB + kl);
          b1.u[vv] = *(const unsigned*)(rowB + 32 + kl);
        }
      }
      v8f acc = {};
      acc = __builtin_amdgcn_wmma_f32_16x16x32_bf16(false, a0.v, false, b0.v, (short)0, acc, false, false);
      acc = __builtin_amdgcn_wmma_f32_16x16x32_bf16(false, a1.v, false, b1.v, (short)0, acc, false, false);
      // scatter D (f32 16x16 layout): address-select, unconditional ds_store
      float* const base = &scorr[0][0];
      #pragma unroll
      for (int r = 0; r < 8; ++r) {
        int m   = r + (hi16 ? 8 : 0);
        int dxp = (n - kR * DIL) - m;               // physical displacement
        int qd; bool valid;
        if constexpr (DIL == 1) {
          qd = dxp;       valid = (dxp >= -kR && dxp <= kR);
        } else {
          qd = dxp >> 1;  valid = ((dxp & 1) == 0) && (qd >= -kR && qd <= kR);
        }
        float* dst = valid ? (base + (m * kN + dyi * kP + qd + kR)) : &strash;
        *dst = acc[r];
      }
    }
    __syncthreads();
  }

  if (tid < 16) {
    const int m = tid;
    const int p = y * kW + x0 + m;
    if (mode == 1) {
      // softmax soft-argmax offsets
      float mx = NEG_INF;
      for (int i = 0; i < kN; ++i) mx = fmaxf(mx, scorr[m][i]);
      float se = 0.f, sy = 0.f, sx = 0.f;
      for (int i = 0; i < kN; ++i) {
        float e = __expf(scorr[m][i] - mx);
        se += e;
        sy += e * (float)(i / kP - kR);
        sx += e * (float)(i % kP - kR);
      }
      out_offy[p] = sy / se * (float)DIL;
      out_offx[p] = sx / se * (float)DIL;
    } else {
      // serial top-20 insertion (stable: ties keep earlier index like lax.top_k)
      float bv[kK]; int bi[kK];
      for (int k = 0; k < kK; ++k) { bv[k] = NEG_INF; bi[k] = 0; }
      for (int i = 0; i < kN; ++i) {
        float v = scorr[m][i];
        if (v > bv[kK - 1]) {
          int j = kK - 1;
          while (j > 0 && v > bv[j - 1]) { bv[j] = bv[j - 1]; bi[j] = bi[j - 1]; --j; }
          bv[j] = v; bi[j] = i;
        }
      }
      for (int k = 0; k < kK; ++k) {
        out_vals[k * kNP + p] = bv[k];
        out_idx [k * kNP + p] = bi[k];
      }
    }
  }
}

// ---------------------------------------------------------------------------
// Long-ref stage 2: corr at fractional offsets == bilinear interp (fixed
// per-pixel weights) of the INTEGER correlation surface. One block per pixel:
// 26x26 integer-corr patch -> 625 interpolated values -> 20-round argmax.
// ---------------------------------------------------------------------------
__global__ __launch_bounds__(256)
void long_stage2_kernel(const float* __restrict__ ft, const float* __restrict__ fr,
                        const float* __restrict__ offy, const float* __restrict__ offx,
                        float* __restrict__ out_vals, int* __restrict__ out_idx) {
  __shared__ float sft[kC];
  __shared__ float sCI[26 * 26];
  __shared__ float sC[kN];
  __shared__ float sRv[256];
  __shared__ int   sRi[256];

  const int tid = threadIdx.x;
  const int p   = blockIdx.x;
  const int y = p / kW, x = p % kW;
  if (tid < kC) sft[tid] = ft[tid * kNP + p];

  float cy = (float)y + offy[p];
  float cx = (float)x + offx[p];
  float byf = floorf(cy), bxf = floorf(cx);
  int by = (int)byf, bx = (int)bxf;
  float fy = cy - byf, fx = cx - bxf;
  __syncthreads();

  for (int i = tid; i < 26 * 26; i += 256) {
    int Y = by - kR + i / 26;
    int X = bx - kR + i % 26;
    float s = 0.f;
    if (Y >= 0 && Y < kH && X >= 0 && X < kW) {
      int base = Y * kW + X;
      #pragma unroll 8
      for (int c = 0; c < kC; ++c) s += sft[c] * fr[c * kNP + base];
    }
    sCI[i] = s;
  }
  __syncthreads();

  float w00 = (1.f - fy) * (1.f - fx), w01 = (1.f - fy) * fx;
  float w10 = fy * (1.f - fx),         w11 = fy * fx;
  for (int i = tid; i < kN; i += 256) {
    int dyi = i / kP, dxi = i - dyi * kP;
    sC[i] = w00 * sCI[dyi * 26 + dxi]       + w01 * sCI[dyi * 26 + dxi + 1]
          + w10 * sCI[(dyi + 1) * 26 + dxi] + w11 * sCI[(dyi + 1) * 26 + dxi + 1];
  }
  __syncthreads();

  for (int r = 0; r < kK; ++r) {
    float bv = NEG_INF; int bi = 0x7FFFFFFF;
    for (int i = tid; i < kN; i += 256) {
      float v = sC[i];
      if (v > bv) { bv = v; bi = i; }
    }
    sRv[tid] = bv; sRi[tid] = bi;
    __syncthreads();
    for (int s = 128; s > 0; s >>= 1) {
      if (tid < s) {
        float ov = sRv[tid + s]; int oi = sRi[tid + s];
        if (ov > sRv[tid] || (ov == sRv[tid] && oi < sRi[tid])) { sRv[tid] = ov; sRi[tid] = oi; }
      }
      __syncthreads();
    }
    if (tid == 0) {
      out_vals[r * kNP + p] = sRv[0];
      out_idx [r * kNP + p] = sRi[0];
      sC[sRi[0]] = NEG_INF;
    }
    __syncthreads();
  }
}

// ---------------------------------------------------------------------------
// Final blend: softmax over 60 candidates, gather quantized colors on the fly
// (bilinear for the long ref, integer+zero-pad for short refs), 32-channel out.
// quantized_r is read strided (::4,::4) directly from the 192x192 source.
// ---------------------------------------------------------------------------
__global__ __launch_bounds__(256)
void blend_kernel(const float* __restrict__ vals, const int* __restrict__ idxb,
                  const float* __restrict__ offy, const float* __restrict__ offx,
                  const float* __restrict__ quant, float* __restrict__ out) {
  int p = blockIdx.x * 256 + threadIdx.x;
  if (p >= kNP) return;
  int y = p / kW, x = p % kW;

  float mx = NEG_INF;
  for (int k = 0; k < kNK; ++k) mx = fmaxf(mx, vals[k * kNP + p]);
  float se = 0.f;
  for (int k = 0; k < kNK; ++k) se += __expf(vals[k * kNP + p] - mx);
  float inv = 1.f / se;

  float acc[kQC];
  #pragma unroll
  for (int c = 0; c < kQC; ++c) acc[c] = 0.f;
  float oy = offy[p], ox = offx[p];

  for (int k = 0; k < kNK; ++k) {
    float wgt = __expf(vals[k * kNP + p] - mx) * inv;
    int ii = idxb[k * kNP + p];
    int dr = ii / kP - kR, dc = ii % kP - kR;
    int ref = k / kK;
    const float* q = quant + (size_t)ref * kQC * 192 * 192;
    if (ref == 0) {
      float py = (float)y + oy + (float)dr;
      float px = (float)x + ox + (float)dc;
      float y0f = floorf(py), x0f = floorf(px);
      int y0 = (int)y0f, x0i = (int)x0f;
      float fy = py - y0f, fxx = px - x0f;
      float w00 = (1.f - fy) * (1.f - fxx) * wgt, w01 = (1.f - fy) * fxx * wgt;
      float w10 = fy * (1.f - fxx) * wgt,         w11 = fy * fxx * wgt;
      bool iy0 = (y0 >= 0 && y0 < kH),         iy1 = (y0 + 1 >= 0 && y0 + 1 < kH);
      bool ix0 = (x0i >= 0 && x0i < kW),       ix1 = (x0i + 1 >= 0 && x0i + 1 < kW);
      #pragma unroll
      for (int c = 0; c < kQC; ++c) {
        const float* qc = q + (size_t)c * 192 * 192;
        float t = 0.f;
        if (iy0 && ix0) t += w00 * qc[(4 * y0) * 192 + 4 * x0i];
        if (iy0 && ix1) t += w01 * qc[(4 * y0) * 192 + 4 * (x0i + 1)];
        if (iy1 && ix0) t += w10 * qc[(4 * (y0 + 1)) * 192 + 4 * x0i];
        if (iy1 && ix1) t += w11 * qc[(4 * (y0 + 1)) * 192 + 4 * (x0i + 1)];
        acc[c] += t;
      }
    } else {
      int Y = y + dr, X = x + dc;
      if (Y >= 0 && Y < kH && X >= 0 && X < kW) {
        #pragma unroll
        for (int c = 0; c < kQC; ++c)
          acc[c] += wgt * q[((size_t)c * 192 + 4 * Y) * 192 + 4 * X];
      }
    }
  }
  #pragma unroll
  for (int c = 0; c < kQC; ++c) out[c * kNP + p] = acc[c];
}

extern "C" void kernel_launch(void* const* d_in, const int* in_sizes, int n_in,
                              void* d_out, int out_size, void* d_ws, size_t ws_size,
                              hipStream_t stream) {
  (void)in_sizes; (void)n_in; (void)out_size; (void)ws_size;
  const float* feats_r = (const float*)d_in[0];   // (3,1,64,48,48)
  const float* feats_t = (const float*)d_in[1];   // (1,64,48,48)
  const float* quant   = (const float*)d_in[2];   // (3,1,32,192,192)

  float* ws   = (float*)d_ws;
  float* offy = ws;                    // 2304
  float* offx = ws + kNP;              // 2304
  float* vals = ws + 2 * kNP;          // 60*2304
  int*   idxb = (int*)(vals + kNK * kNP);  // 60*2304

  const int FSZ = kC * kNP;
  dim3 cgrid(3, 48);

  // gaps = 60 - {40,50,55} = {20,10,5}; only 20 > DIL_INT=16 -> ref0 long, dirate=min(4,20/16+1)=2
  corr_topk_kernel<2><<<cgrid, 256, 0, stream>>>(feats_t, feats_r + 0 * FSZ, 1,
                                                 nullptr, nullptr, offy, offx);
  long_stage2_kernel<<<kNP, 256, 0, stream>>>(feats_t, feats_r + 0 * FSZ, offy, offx,
                                              vals, idxb);
  corr_topk_kernel<1><<<cgrid, 256, 0, stream>>>(feats_t, feats_r + 1 * FSZ, 0,
                                                 vals + 1 * kK * kNP, idxb + 1 * kK * kNP,
                                                 nullptr, nullptr);
  corr_topk_kernel<1><<<cgrid, 256, 0, stream>>>(feats_t, feats_r + 2 * FSZ, 0,
                                                 vals + 2 * kK * kNP, idxb + 2 * kK * kNP,
                                                 nullptr, nullptr);
  blend_kernel<<<(kNP + 255) / 256, 256, 0, stream>>>(vals, idxb, offy, offx,
                                                      quant, (float*)d_out);
}
